// Gru4Rec_83562883711518
// MI455X (gfx1250) — matrix operand order repaired and checked by reading, not compiled, not run on hardware
//
#include <hip/hip_runtime.h>
#include <hip/hip_bf16.h>

// ---------------------------------------------------------------------------
// GRU4Rec forward on MI455X (gfx1250), wave32 + bf16 WMMA (f32 accumulate).
// Round 4: grid-order fix for the dominant GEMM (A=256MB > L2=192MB, so the
// 5 N-block-columns must be dispatched adjacently to reuse each A tile in L2;
// otherwise A streams from HBM 5x = ~1.3GB instead of once).
// ---------------------------------------------------------------------------

typedef __attribute__((ext_vector_type(16))) __bf16 v16bf;
typedef __attribute__((ext_vector_type(8)))  __bf16 v8bf;
typedef __attribute__((ext_vector_type(8)))  float  v8f;
// 16B-alignment-reduced v8f for global loads (lda=100 rows are only 16B aligned)
typedef v8f __attribute__((aligned(16))) v8fa;

__device__ __forceinline__ v16bf make_frag(const __bf16* lo, const __bf16* hi) {
    v8bf a = *(const v8bf*)lo;
    v8bf b = *(const v8bf*)hi;
    v16bf r;
#pragma unroll
    for (int i = 0; i < 8; ++i) { r[i] = a[i]; r[i + 8] = b[i]; }
    return r;
}

// ===========================================================================
// Tiled GEMM: C[M,N] = act( A[M,K](f32) * B[K,N](f32) + bias[N] )
// REQUIREMENT: M % 128 == 0 (true for all call sites: 12800, 256).
// Block tile 128(M) x 64(N), 8 waves, K-step 32, bf16 WMMA, f32 accum.
// n_fastest=1: blockIdx.x indexes N (use when A >> L2 and N-columns must
//              share A tiles);  n_fastest=0: blockIdx.x indexes M.
// ===========================================================================
#define GEMM_TB 256
#define GEMM_BM 128
#define GEMM_BN 64
#define GEMM_KS 32
#define GEMM_AST 40  // A LDS k-stride (32+8 pad) -> conflict-free b128 frag reads
#define GEMM_BST 40  // B LDS k-stride (col-major [n][k])

__device__ __forceinline__ void load_a_tile(const float* __restrict__ A, int lda,
                                            int K, int m_base, int k0, int tid,
                                            v8f* r) {
    const int row = tid >> 1;
    const int cb  = (tid & 1) * 16;
    const float* ap = A + (size_t)(m_base + row) * lda + (k0 + cb);
    if (k0 + cb + 16 <= K) {              // fast: 16B-aligned vector loads
        r[0] = *(const v8fa*)(ap + 0);
        r[1] = *(const v8fa*)(ap + 8);
    } else {                               // K tail only
#pragma unroll
        for (int j = 0; j < 8; ++j) {
            r[0][j] = (k0 + cb + j     < K) ? ap[j]     : 0.0f;
            r[1][j] = (k0 + cb + j + 8 < K) ? ap[j + 8] : 0.0f;
        }
    }
}

__device__ __forceinline__ void store_a_lds(__bf16* Alds, int tid, const v8f* r) {
    const int row = tid >> 1;
    const int cb  = (tid & 1) * 16;
    const v8bf b0 = __builtin_convertvector(r[0], v8bf);  // packed cvt_pk_bf16
    const v8bf b1 = __builtin_convertvector(r[1], v8bf);
    v8bf* dst = (v8bf*)&Alds[row * GEMM_AST + cb];        // 16B aligned
    dst[0] = b0;
    dst[1] = b1;
}

// B tile: thread -> (n = tid&63, 8 consecutive k at kb=(tid>>6)*8).
// Global reads coalesced across lanes (consecutive n); LDS write is one b128.
__device__ __forceinline__ void load_b_tile(const float* __restrict__ Bm, int ldb,
                                            int N, int K, int n_base, int k0,
                                            int tid, v8f* r) {
    const int n  = tid & 63;
    const int kb = (tid >> 6) * 8;
    const int gn = n_base + n;
    const float* bp = Bm + (size_t)(k0 + kb) * ldb + gn;
    if (gn < N && (k0 + kb + 8) <= K) {   // fast: 8 coalesced b32 loads
#pragma unroll
        for (int j = 0; j < 8; ++j) (*r)[j] = bp[(size_t)j * ldb];
    } else {                               // K tail / partial-N column
#pragma unroll
        for (int j = 0; j < 8; ++j)
            (*r)[j] = (gn < N && (k0 + kb + j) < K) ? bp[(size_t)j * ldb] : 0.0f;
    }
}

__device__ __forceinline__ void store_b_lds(__bf16* Blds, int tid, const v8f* r) {
    const int n  = tid & 63;
    const int kb = (tid >> 6) * 8;
    const v8bf b = __builtin_convertvector(*r, v8bf);
    *(v8bf*)&Blds[n * GEMM_BST + kb] = b;                 // one ds_store_b128
}

__global__ __launch_bounds__(GEMM_TB)
void gemm_bf16_wmma(const float* __restrict__ A, const float* __restrict__ Bm,
                    const float* __restrict__ bias, float* __restrict__ C,
                    int M, int N, int K, int lda, int ldb, int ldc,
                    int act_tanh, int n_fastest)
{
    __shared__ __align__(16) __bf16 Alds[GEMM_BM * GEMM_AST];
    __shared__ __align__(16) __bf16 Blds[GEMM_BN * GEMM_BST];

    const int tid  = threadIdx.x;
    const int wave = tid >> 5;
    const int lane = tid & 31;
    const int lr   = lane & 15;
    const int lhi  = lane >> 4;
    const int bm = n_fastest ? blockIdx.y : blockIdx.x;
    const int bn = n_fastest ? blockIdx.x : blockIdx.y;
    const int m_base = bm * GEMM_BM;
    const int n_base = bn * GEMM_BN;

    v8f acc[4];
#pragma unroll
    for (int i = 0; i < 4; ++i)
#pragma unroll
        for (int j = 0; j < 8; ++j) acc[i][j] = 0.0f;

    const int ksteps = (K + GEMM_KS - 1) / GEMM_KS;

    v8f aR[2], bR;
    load_a_tile(A, lda, K, m_base, 0, tid, aR);
    load_b_tile(Bm, ldb, N, K, n_base, 0, tid, &bR);

    for (int kk = 0; kk < ksteps; ++kk) {
        store_a_lds(Alds, tid, aR);
        store_b_lds(Blds, tid, &bR);
        __syncthreads();

        // prefetch next tile into registers while WMMAs run from LDS
        if (kk + 1 < ksteps) {
            load_a_tile(A, lda, K, m_base, (kk + 1) * GEMM_KS, tid, aR);
            load_b_tile(Bm, ldb, N, K, n_base, (kk + 1) * GEMM_KS, tid, &bR);
        }

        const __bf16* apf = &Alds[(wave * 16 + lr) * GEMM_AST + lhi * 8];
        const v16bf afrag = make_frag(apf, apf + 16);
        v16bf bfr[4];
#pragma unroll
        for (int nt = 0; nt < 4; ++nt) {
            const __bf16* bpf = &Blds[(nt * 16 + lr) * GEMM_BST + lhi * 8];
            bfr[nt] = make_frag(bpf, bpf + 16);
        }
#pragma unroll
        for (int nt = 0; nt < 4; ++nt)
            acc[nt] = __builtin_amdgcn_wmma_f32_16x16x32_bf16(
                false, afrag, false, bfr[nt], (short)0, acc[nt], false, false);
        __syncthreads();
    }

    // epilogue (rows always valid: M % 128 == 0)
#pragma unroll
    for (int nt = 0; nt < 4; ++nt) {
        const int col = n_base + nt * 16 + lr;
        if (col < N) {
            const float bv = bias ? bias[col] : 0.0f;
#pragma unroll
            for (int i = 0; i < 8; ++i) {
                const int row = m_base + wave * 16 + lhi * 8 + i;
                float v = acc[nt][i] + bv;
                if (act_tanh) v = tanhf(v);
                C[(size_t)row * ldc + col] = v;
            }
        }
    }
}

// ===========================================================================
// Recurrent kernel -> bf16 B-fragment layout [col 0..303][k 0..127], padded.
// ===========================================================================
__global__ __launch_bounds__(256)
void convert_wr(const float* __restrict__ grk, __bf16* __restrict__ out)
{
    const int idx = blockIdx.x * 256 + threadIdx.x;
    if (idx >= 304 * 128) return;
    const int col = idx >> 7;
    const int k   = idx & 127;
    const float v = (col < 300 && k < 100) ? grk[k * 300 + col] : 0.0f;
    out[idx] = (__bf16)v;
}

// ===========================================================================
// GRU scan: each block owns 16 batch rows for all 50 steps; h lives in LDS.
// hproj(16x300) via WMMA against L2-resident bf16 Wr. No cross-block sync.
// ===========================================================================
__global__ __launch_bounds__(256)
void gru_scan(const float* __restrict__ xproj,      // (12800 x 300) f32
              const __bf16* __restrict__ wrbf,      // [304][128] bf16
              const float* __restrict__ rbias,      // (300,)
              float* __restrict__ hfinal)           // (256 x 100) f32
{
    __shared__ __align__(16) __bf16 hbf[16 * 128];
    __shared__ float  hf [16 * 100];
    __shared__ float  hp [16 * 304];
    __shared__ float  rb [304];

    const int tid  = threadIdx.x;
    const int wave = tid >> 5;
    const int lane = tid & 31;
    const int lr   = lane & 15;
    const int lhi  = lane >> 4;
    const int blk  = blockIdx.x;

    for (int i = tid; i < 16 * 128; i += 256) hbf[i] = (__bf16)0.0f;
    for (int i = tid; i < 16 * 100; i += 256) hf[i] = 0.0f;
    for (int i = tid; i < 304;      i += 256) rb[i] = (i < 300) ? rbias[i] : 0.0f;
    __syncthreads();

    for (int t = 0; t < 50; ++t) {
        // ---- hproj = h @ Wr : 19 N-tiles, K=128 (zero-padded) --------------
        for (int nt = wave; nt < 19; nt += 8) {      // wave-uniform branch
            v16bf a[4], b[4];
#pragma unroll
            for (int kk = 0; kk < 4; ++kk) {
                const __bf16* apf = &hbf[lr * 128 + kk * 32 + lhi * 8];
                a[kk] = make_frag(apf, apf + 16);
                const __bf16* bpf = &wrbf[(nt * 16 + lr) * 128 + kk * 32 + lhi * 8];
                b[kk] = make_frag(bpf, bpf + 16);
            }
            v8f acc;
#pragma unroll
            for (int j = 0; j < 8; ++j) acc[j] = 0.0f;
#pragma unroll
            for (int kk = 0; kk < 4; ++kk)
                acc = __builtin_amdgcn_wmma_f32_16x16x32_bf16(
                    false, a[kk], false, b[kk], (short)0, acc, false, false);
            const int col = nt * 16 + lr;            // <= 303
#pragma unroll
            for (int i = 0; i < 8; ++i)
                hp[(lhi * 8 + i) * 304 + col] = acc[i];
        }
        __syncthreads();

        // ---- gates (elementwise per (row,g)) ------------------------------
        for (int i = tid; i < 1600; i += 256) {
            const int row = i / 100;
            const int g   = i % 100;
            const int batch = blk * 16 + row;
            const float* xp = xproj + (size_t)(batch * 50 + t) * 300;
            const float xz = xp[g];
            const float xr = xp[100 + g];
            const float xh = xp[200 + g];
            const float hz = hp[row * 304 + g]       + rb[g];
            const float hr = hp[row * 304 + 100 + g] + rb[100 + g];
            const float hh = hp[row * 304 + 200 + g] + rb[200 + g];
            const float z  = 1.0f / (1.0f + __expf(-(xz + hz)));
            const float r  = 1.0f / (1.0f + __expf(-(xr + hr)));
            const float cand = tanhf(xh + r * hh);
            const float hold = hf[row * 100 + g];
            const float hnew = z * hold + (1.0f - z) * cand;
            hf[row * 100 + g]  = hnew;
            hbf[row * 128 + g] = (__bf16)hnew;
        }
        __syncthreads();
    }

    for (int i = tid; i < 1600; i += 256) {
        const int row = i / 100;
        const int g   = i % 100;
        hfinal[(size_t)(blk * 16 + row) * 100 + g] = hf[row * 100 + g];
    }
}

// ===========================================================================
// Host launcher
// ===========================================================================
extern "C" void kernel_launch(void* const* d_in, const int* in_sizes, int n_in,
                              void* d_out, int out_size, void* d_ws, size_t ws_size,
                              hipStream_t stream) {
    (void)in_sizes; (void)n_in; (void)out_size; (void)ws_size;

    const float* inputs = (const float*)d_in[0];   // (256,50,5000)
    const float* gk     = (const float*)d_in[1];   // (5000,300)
    const float* grk    = (const float*)d_in[2];   // (100,300)
    const float* gib    = (const float*)d_in[3];   // (300,)
    const float* grb    = (const float*)d_in[4];   // (300,)
    const float* w1     = (const float*)d_in[5];   // (100,5000)
    const float* b1     = (const float*)d_in[6];   // (5000,)
    const float* w2     = (const float*)d_in[7];   // (5000,5000)
    const float* b2     = (const float*)d_in[8];   // (5000,)
    float* out = (float*)d_out;                    // (256,5000)

    char* ws = (char*)d_ws;
    size_t off = 0;
    float* xproj = (float*)(ws + off); off += (size_t)12800 * 300 * sizeof(float);
    float* hfin  = (float*)(ws + off); off += (size_t)256 * 100 * sizeof(float);
    float* mid   = (float*)(ws + off); off += (size_t)256 * 5000 * sizeof(float);
    off = (off + 255) & ~(size_t)255;
    __bf16* wrbf = (__bf16*)(ws + off);            // 304*128*2 B

    convert_wr<<<(304 * 128 + 255) / 256, 256, 0, stream>>>(grk, wrbf);

    // xproj = inputs @ gru_kernel + input_bias   (M=12800, N=300, K=5000)
    // A (256MB) > L2: N-blocks fastest so each A tile is reused from L2.
    gemm_bf16_wmma<<<dim3(5, 100), GEMM_TB, 0, stream>>>(
        inputs, gk, gib, xproj, 12800, 300, 5000, 5000, 300, 300, 0, 1);

    // 50-step GRU scan -> h_final
    gru_scan<<<16, 256, 0, stream>>>(xproj, wrbf, grb, hfin);

    // mid = tanh(h_final @ w1 + b1)              (M=256, N=5000, K=100)
    gemm_bf16_wmma<<<dim3(2, 79), GEMM_TB, 0, stream>>>(
        hfin, w1, b1, mid, 256, 5000, 100, 100, 5000, 5000, 1, 0);

    // out = mid @ w2 + b2                        (M=256, N=5000, K=5000)
    // B (w2, 100MB) read once with M-blocks fastest; A (5MB) stays in L2.
    gemm_bf16_wmma<<<dim3(2, 79), GEMM_TB, 0, stream>>>(
        mid, w2, b2, out, 256, 5000, 5000, 5000, 5000, 5000, 0, 0);
}
